// EMASpitDelta_35588099014895
// MI455X (gfx1250) — compile-verified, matrix operand order checked by reading, not run on hardware
//
#include <hip/hip_runtime.h>
#include <hip/hip_bf16.h>

#define B_N 128
#define L_N 4096
#define H_N 64
#define V_N 64
#define HALF_N 32
#define LM1 4095

typedef __attribute__((ext_vector_type(16))) __bf16 v16bf;
typedef __attribute__((ext_vector_type(8)))  float  v8f;
typedef __attribute__((ext_vector_type(2)))  float  v2f;

#define WMMA_BF16(A, Bf, C) \
  __builtin_amdgcn_wmma_f32_16x16x32_bf16(false, (A), false, (Bf), (short)0, (C), false, false)
#define WMMA_F32(A, Bf, C) \
  __builtin_amdgcn_wmma_f32_16x16x4_f32(false, (A), false, (Bf), (short)0, (C), false, false)

// Load a 16x32 bf16 WMMA fragment (A layout; B uses the same pattern on a
// transposed [N][K] tile). ISA 7.12.2: lane half (hi) selects K+8; VGPR v
// holds K pair {2v, 2v+1} (v<4) or {16+2(v-4), ...} (v>=4).
__device__ __forceinline__ v16bf ldfrag(const __bf16* base, int rc, int stride,
                                        int koff, int hi) {
  union { v16bf v; unsigned u[8]; } f;
  const __bf16* p = base + rc * stride + koff + hi * 8;
#pragma unroll
  for (int q = 0; q < 8; ++q) {
    int kk = (q < 4) ? (2 * q) : (16 + 2 * (q - 4));
    f.u[q] = *(const unsigned*)(p + kk);
  }
  return f.v;
}

// ---------------------------------------------------------------------------
// Kernel 0: one-time weight prep (runs once, 1 block).
//   w1tB[n][k] = bf16(W1[k][n])   (128x64)
//   w2tB[n][k] = bf16(W2[k][n])   (64x128)
//   wstB[n][k] = Ws[k][n]         (32x64 f32)
//   wetB[n][k] = We[k][n]         (32x64 f32)
//   embB       = bf16(embed)      (64x64)
// ---------------------------------------------------------------------------
__global__ __launch_bounds__(256) void prep_kernel(
    const float* __restrict__ embed, const float* __restrict__ W1,
    const float* __restrict__ W2, const float* __restrict__ Ws,
    const float* __restrict__ We, __bf16* __restrict__ w1tB,
    __bf16* __restrict__ w2tB, float* __restrict__ wstB,
    float* __restrict__ wetB, __bf16* __restrict__ embB) {
  int tid = threadIdx.x;
  for (int idx = tid; idx < 8192; idx += 256) {
    int k = idx >> 7, n = idx & 127;
    w1tB[n * 64 + k] = (__bf16)W1[idx];
  }
  for (int idx = tid; idx < 8192; idx += 256) {
    int k = idx >> 6, n = idx & 63;
    w2tB[n * 128 + k] = (__bf16)W2[idx];
  }
  for (int idx = tid; idx < 2048; idx += 256) {
    int k = idx >> 5, n = idx & 31;
    wstB[n * 64 + k] = Ws[idx];
    wetB[n * 64 + k] = We[idx];
  }
  for (int idx = tid; idx < 4096; idx += 256)
    embB[idx] = (__bf16)embed[idx];
}

// ---------------------------------------------------------------------------
// Kernel 1: per-token  embed-gather -> MLP (WMMA bf16/f32acc) -> LN ->
//           projections (WMMA f32 16x16x4) -> normalize -> store
// 64 tokens per block (128 threads = 4 waves, 16 tokens per wave)
// ---------------------------------------------------------------------------
__global__ __launch_bounds__(128) void token_kernel(
    const int* __restrict__ seq, const float* __restrict__ embed,
    const __bf16* __restrict__ w1tB, const __bf16* __restrict__ w2tB,
    const float* __restrict__ wstB, const float* __restrict__ wetB,
    const __bf16* __restrict__ embB,
    const float* __restrict__ b1, const float* __restrict__ b2,
    const float* __restrict__ gamma, const float* __restrict__ beta,
    const float* __restrict__ bs, const float* __restrict__ be,
    float* __restrict__ hsB, float* __restrict__ heB,
    float* __restrict__ ksB, float* __restrict__ keB,
    float* __restrict__ qsB, float* __restrict__ qeB) {
  __shared__ __align__(16) char smem[60160];
  // region0 [0,16384): W1^T bf16[128][64]; later Ws^T/We^T f32[32][64] each
  __bf16* sW1t = (__bf16*)(smem);
  __bf16* sW2t = (__bf16*)(smem + 16384);   // W2^T bf16[64][128]
  float*  sWst = (float*)(smem);            // [32][64] f32 (alias region0)
  float*  sWet = (float*)(smem + 8192);     // [32][64] f32
  int*    sTok = (int*)(smem + 57344);
  float*  sB1  = (float*)(smem + 57600);
  float*  sB2  = (float*)(smem + 58112);
  float*  sG   = (float*)(smem + 58368);
  float*  sBt  = (float*)(smem + 58624);
  float*  sBs  = (float*)(smem + 58880);
  float*  sBe  = (float*)(smem + 59008);

  const int tid  = threadIdx.x;
  const int wv   = tid >> 5;
  const int lane = tid & 31;
  const int g0   = blockIdx.x * 64;   // flat token base (b*L + l)

  // ---- stage weights (b128 copies of pre-converted data) -------------------
  if (tid < 64) sTok[tid] = seq[g0 + tid];
#pragma unroll
  for (int idx = tid; idx < 1024; idx += 128)
    ((uint4*)sW1t)[idx] = ((const uint4*)w1tB)[idx];
#pragma unroll
  for (int idx = tid; idx < 1024; idx += 128)
    ((uint4*)sW2t)[idx] = ((const uint4*)w2tB)[idx];
  if (tid < 128) sB1[tid] = b1[tid];
  if (tid < 64) { sB2[tid] = b2[tid]; sG[tid] = gamma[tid]; sBt[tid] = beta[tid]; }
  if (tid < 32) { sBs[tid] = bs[tid]; sBe[tid] = be[tid]; }
  __syncthreads();

  // ---- build A tiles (16x64 bf16 per wave) from bf16 embedding gather ------
#pragma unroll
  for (int idx = tid; idx < 1024; idx += 128) {
    int tl = idx >> 4, dq = (idx & 15) << 2;       // 4 bf16 (8B) per copy
    __bf16* Aw = (__bf16*)(smem + 32768 + (tl >> 4) * 6144);
    *(uint2*)(Aw + (tl & 15) * 64 + dq) = *(const uint2*)(embB + sTok[tl] * 64 + dq);
  }
  __syncthreads();

  const int hi = lane >> 4, n16 = lane & 15;
  __bf16* Aw  = (__bf16*)(smem + 32768 + wv * 6144);
  __bf16* Y1w = (__bf16*)(smem + 32768 + wv * 6144 + 2048);  // 16x128 bf16
  float*  Xw  = (float*)(Y1w);                               // alias: 16x64 f32

  // ---- GEMM1: (16x64)@(64x128), relu, -> Y1 bf16 ---------------------------
  v16bf a0 = ldfrag(Aw, n16, 64, 0, hi);
  v16bf a1 = ldfrag(Aw, n16, 64, 32, hi);
#pragma unroll
  for (int nt = 0; nt < 8; ++nt) {
    v8f acc = {};
    acc = WMMA_BF16(a0, ldfrag(sW1t + nt * 16 * 64, n16, 64, 0, hi), acc);
    acc = WMMA_BF16(a1, ldfrag(sW1t + nt * 16 * 64, n16, 64, 32, hi), acc);
    int n = nt * 16 + n16;
    float bias = sB1[n];
#pragma unroll
    for (int r = 0; r < 8; ++r) {
      int m = r + hi * 8;
      float v = acc[r] + bias;
      Y1w[m * 128 + n] = (__bf16)(v > 0.f ? v : 0.f);
    }
  }
  __syncthreads();   // all waves finished reading W1t -> region0 reusable

  // stage pre-transposed Ws^T/We^T (f32) over W1t region
#pragma unroll
  for (int idx = tid; idx < 512; idx += 128)
    ((uint4*)sWst)[idx] = ((const uint4*)wstB)[idx];
#pragma unroll
  for (int idx = tid; idx < 512; idx += 128)
    ((uint4*)sWet)[idx] = ((const uint4*)wetB)[idx];

  // ---- GEMM2: (16x128)@(128x64) + b2 + residual h0 -> X f32 ----------------
  v16bf y0 = ldfrag(Y1w, n16, 128, 0, hi);
  v16bf y1 = ldfrag(Y1w, n16, 128, 32, hi);
  v16bf y2 = ldfrag(Y1w, n16, 128, 64, hi);
  v16bf y3 = ldfrag(Y1w, n16, 128, 96, hi);
#pragma unroll
  for (int nt = 0; nt < 4; ++nt) {
    v8f acc = {};
    acc = WMMA_BF16(y0, ldfrag(sW2t + nt * 16 * 128, n16, 128, 0, hi), acc);
    acc = WMMA_BF16(y1, ldfrag(sW2t + nt * 16 * 128, n16, 128, 32, hi), acc);
    acc = WMMA_BF16(y2, ldfrag(sW2t + nt * 16 * 128, n16, 128, 64, hi), acc);
    acc = WMMA_BF16(y3, ldfrag(sW2t + nt * 16 * 128, n16, 128, 96, hi), acc);
    int n = nt * 16 + n16;
    float bias = sB2[n];
#pragma unroll
    for (int r = 0; r < 8; ++r) {
      int m = r + hi * 8;
      int tok = sTok[wv * 16 + m];
      Xw[m * 64 + n] = acc[r] + bias + embed[tok * 64 + n];   // residual (L2, f32)
    }
  }
  __syncthreads();   // Wst/Wet staged by all threads; X tiles final

  // ---- LayerNorm (f32, 2 threads per token, all within own wave) -----------
  {
    const int tl = tid >> 1, half = tid & 1;
    float* Xt = (float*)(smem + 32768 + (tl >> 4) * 6144 + 2048) + (tl & 15) * 64;
    float mu = 0.f;
#pragma unroll
    for (int i = 0; i < 64; ++i) mu += Xt[i];
    mu *= (1.f / 64.f);
    float var = 0.f;
#pragma unroll
    for (int i = 0; i < 64; ++i) { float d = Xt[i] - mu; var = fmaf(d, d, var); }
    float rstd = rsqrtf(var * (1.f / 64.f) + 1e-5f);
    // wave-synchronous: all lanes' loads above precede these stores
#pragma unroll
    for (int jj = 0; jj < 32; ++jj) {
      int i = half * 32 + jj;
      Xt[i] = (Xt[i] - mu) * rstd * sG[i] + sBt[i];
    }
  }

  // ---- projections via f32 WMMA 16x16x4: h(16x64) @ Ws/We(64x32) -----------
  {
    float* Xf  = (float*)(smem + 32768 + wv * 6144 + 2048);  // h tile, 16x64
    float* hsT = (float*)(smem + 32768 + wv * 6144);         // out 16x32 (over A)
    float* heT = Xf;                                         // out 16x32 (over X)
    v2f af[16];
#pragma unroll
    for (int kc = 0; kc < 16; ++kc)
      af[kc] = *(const v2f*)(Xf + n16 * 64 + kc * 4 + hi * 2);
    v8f aS0 = {}, aS1 = {}, aE0 = {}, aE1 = {};
#pragma unroll
    for (int kc = 0; kc < 16; ++kc) {
      v2f bS0 = *(const v2f*)(sWst + n16 * 64 + kc * 4 + hi * 2);
      v2f bS1 = *(const v2f*)(sWst + (16 + n16) * 64 + kc * 4 + hi * 2);
      v2f bE0 = *(const v2f*)(sWet + n16 * 64 + kc * 4 + hi * 2);
      v2f bE1 = *(const v2f*)(sWet + (16 + n16) * 64 + kc * 4 + hi * 2);
      aS0 = WMMA_F32(af[kc], bS0, aS0);
      aS1 = WMMA_F32(af[kc], bS1, aS1);
      aE0 = WMMA_F32(af[kc], bE0, aE0);
      aE1 = WMMA_F32(af[kc], bE1, aE1);
    }
#pragma unroll
    for (int r = 0; r < 8; ++r) {
      int m = r + hi * 8;
      hsT[m * 32 + n16]      = aS0[r] + sBs[n16];
      hsT[m * 32 + 16 + n16] = aS1[r] + sBs[16 + n16];
      heT[m * 32 + n16]      = aE0[r] + sBe[n16];
      heT[m * 32 + 16 + n16] = aE1[r] + sBe[16 + n16];
    }
  }

  // ---- normalize + store (2 threads per token, own wave data) --------------
  {
    const int tl = tid >> 1, half = tid & 1;
    float* hsRow = (float*)(smem + 32768 + (tl >> 4) * 6144) + (tl & 15) * 32;
    float* heRow = (float*)(smem + 32768 + (tl >> 4) * 6144 + 2048) + (tl & 15) * 32;
    float ssS = 0.f, ssE = 0.f;
#pragma unroll
    for (int j = 0; j < 32; ++j) {
      float a = hsRow[j], bv = heRow[j];
      ssS = fmaf(a, a, ssS);
      ssE = fmaf(bv, bv, ssE);
    }
    float rnS = 1.f / fmaxf(sqrtf(ssS), 1e-12f);
    float rnE = 1.f / fmaxf(sqrtf(ssE), 1e-12f);

    int g = g0 + tl, b = g >> 12, l = g & 4095;
    if (l < LM1) {
      size_t base = ((size_t)b * LM1 + l) * 32 + half * 16;
#pragma unroll
      for (int jj = 0; jj < 16; ++jj) {
        int j = half * 16 + jj;
        float a = hsRow[j], bv = heRow[j];
        hsB[base + jj] = a;
        heB[base + jj] = bv;
        ksB[base + jj] = a * rnS;
        keB[base + jj] = bv * rnE;
      }
    } else {
      int qb = b * 32 + half * 16;
#pragma unroll
      for (int jj = 0; jj < 16; ++jj) {
        int j = half * 16 + jj;
        qsB[qb + jj] = hsRow[j] * rnS;
        qeB[qb + jj] = heRow[j] * rnE;
      }
    }
  }
}

// ---------------------------------------------------------------------------
// Kernel 2: sequential delta-rule EMA scan.  One wave per (batch, {s,e}).
// M (32x32) register-resident: lane i = row i, packed as 16 float2 along j
// (V_PK_FMA_F32 path).  k[j] broadcast via readlane; loads pipelined.
// ---------------------------------------------------------------------------
__device__ __forceinline__ float rl(float v, int l) {
  return __builtin_bit_cast(float,
      __builtin_amdgcn_readlane(__builtin_bit_cast(int, v), l));
}

__global__ __launch_bounds__(256) void scan_kernel(
    const float* __restrict__ hsB, const float* __restrict__ heB,
    const float* __restrict__ ksB, const float* __restrict__ keB,
    const float* __restrict__ qsB, const float* __restrict__ qeB,
    float* __restrict__ rB) {
  int w    = (blockIdx.x * blockDim.x + threadIdx.x) >> 5;   // 0..255
  int lane = threadIdx.x & 31;
  int b = w >> 1, e = w & 1;
  const float* Hh = e ? heB : hsB;
  const float* Kk = e ? keB : ksB;
  size_t base = (size_t)b * LM1 * 32 + lane;

  v2f M2[16];
#pragma unroll
  for (int p = 0; p < 16; ++p) M2[p] = (v2f){0.f, 0.f};

  const float cb   = 1.f - 0.95f;
  const float invL = 1.f / (float)L_N;

  float kv = Kk[base];
  float hv = Hh[base];
  for (int t = 0; t < LM1; ++t) {
    int tn = (t + 1 < LM1) ? (t + 1) : (LM1 - 1);
    float kn = Kk[base + (size_t)tn * 32];         // pipelined next-step load
    float hn = Hh[base + (size_t)tn * 32];
    __builtin_prefetch(&Kk[base + (size_t)(t + 16) * 32], 0, 0);
    __builtin_prefetch(&Hh[base + (size_t)(t + 16) * 32], 0, 0);

    v2f kp[16];
#pragma unroll
    for (int p = 0; p < 16; ++p)
      kp[p] = (v2f){rl(kv, 2 * p), rl(kv, 2 * p + 1)};

    v2f a0 = (v2f){0.f, 0.f}, a1 = a0, a2 = a0, a3 = a0;
#pragma unroll
    for (int p = 0; p < 4; ++p) {
      a0 = __builtin_elementwise_fma(M2[p],      kp[p],      a0);
      a1 = __builtin_elementwise_fma(M2[4 + p],  kp[4 + p],  a1);
      a2 = __builtin_elementwise_fma(M2[8 + p],  kp[8 + p],  a2);
      a3 = __builtin_elementwise_fma(M2[12 + p], kp[12 + p], a3);
    }
    float vp = ((a0.x + a0.y) + (a1.x + a1.y)) + ((a2.x + a2.y) + (a3.x + a3.y));
    float coef  = e ? (cb * ((float)(t + 1) * invL)) : cb;
    float delta = coef * (hv - vp);
    v2f d2 = (v2f){delta, delta};
#pragma unroll
    for (int p = 0; p < 16; ++p)
      M2[p] = __builtin_elementwise_fma(d2, kp[p], M2[p]);
    kv = kn;
    hv = hn;
  }

  float qv = (e ? qeB : qsB)[b * 32 + lane];
  v2f r0 = (v2f){0.f, 0.f}, r1 = r0, r2 = r0, r3 = r0;
#pragma unroll
  for (int p = 0; p < 4; ++p) {
    v2f q0 = (v2f){rl(qv, 2 * p), rl(qv, 2 * p + 1)};
    v2f q1 = (v2f){rl(qv, 8 + 2 * p), rl(qv, 8 + 2 * p + 1)};
    v2f q2 = (v2f){rl(qv, 16 + 2 * p), rl(qv, 16 + 2 * p + 1)};
    v2f q3 = (v2f){rl(qv, 24 + 2 * p), rl(qv, 24 + 2 * p + 1)};
    r0 = __builtin_elementwise_fma(M2[p],      q0, r0);
    r1 = __builtin_elementwise_fma(M2[4 + p],  q1, r1);
    r2 = __builtin_elementwise_fma(M2[8 + p],  q2, r2);
    r3 = __builtin_elementwise_fma(M2[12 + p], q3, r3);
  }
  rB[b * 64 + e * 32 + lane] =
      ((r0.x + r0.y) + (r1.x + r1.y)) + ((r2.x + r2.y) + (r3.x + r3.y));
}

// ---------------------------------------------------------------------------
// Kernel 3: head  out = (r @ Wrp + brp) @ Wout + bout   (tiny)
// ---------------------------------------------------------------------------
__global__ __launch_bounds__(64) void head_kernel(
    const float* __restrict__ rB, const float* __restrict__ Wrp,
    const float* __restrict__ brp, const float* __restrict__ Wout,
    const float* __restrict__ bout, float* __restrict__ out) {
  __shared__ float rsh[64];
  __shared__ float z[64];
  int t = threadIdx.x, b = blockIdx.x;
  rsh[t] = rB[b * 64 + t];
  __syncthreads();
  float acc = brp[t];
#pragma unroll
  for (int j = 0; j < 64; ++j) acc = fmaf(rsh[j], Wrp[j * 64 + t], acc);
  z[t] = acc;
  __syncthreads();
  float o = bout[t];
#pragma unroll
  for (int k = 0; k < 64; ++k) o = fmaf(z[k], Wout[k * 64 + t], o);
  out[b * 64 + t] = o;
}

extern "C" void kernel_launch(void* const* d_in, const int* in_sizes, int n_in,
                              void* d_out, int out_size, void* d_ws, size_t ws_size,
                              hipStream_t stream) {
  (void)in_sizes; (void)n_in; (void)out_size; (void)ws_size;
  const int*   seq   = (const int*)d_in[0];
  const float* embed = (const float*)d_in[1];
  const float* W1    = (const float*)d_in[2];
  const float* b1    = (const float*)d_in[3];
  const float* W2    = (const float*)d_in[4];
  const float* b2    = (const float*)d_in[5];
  const float* gamma = (const float*)d_in[6];
  const float* beta  = (const float*)d_in[7];
  const float* Ws    = (const float*)d_in[8];
  const float* bs    = (const float*)d_in[9];
  const float* We    = (const float*)d_in[10];
  const float* be    = (const float*)d_in[11];
  const float* Wrp   = (const float*)d_in[12];
  const float* brp   = (const float*)d_in[13];
  const float* Wout  = (const float*)d_in[14];
  const float* bout  = (const float*)d_in[15];

  float* ws = (float*)d_ws;
  size_t N1 = (size_t)B_N * LM1 * 32;
  float* hsB = ws;
  float* heB = ws + N1;
  float* ksB = ws + 2 * N1;
  float* keB = ws + 3 * N1;
  float* qsB = ws + 4 * N1;
  float* qeB = qsB + B_N * 32;
  float* rB  = qeB + B_N * 32;
  float* extra = rB + B_N * 64;
  __bf16* w1tB = (__bf16*)extra;             // 8192 bf16
  __bf16* w2tB = (__bf16*)(extra + 4096);    // 8192 bf16
  float*  wstB = extra + 8192;               // 2048 f32
  float*  wetB = extra + 10240;              // 2048 f32
  __bf16* embB = (__bf16*)(extra + 12288);   // 4096 bf16

  prep_kernel<<<1, 256, 0, stream>>>(embed, W1, W2, Ws, We,
                                     w1tB, w2tB, wstB, wetB, embB);
  token_kernel<<<(B_N * L_N) / 64, 128, 0, stream>>>(
      seq, embed, w1tB, w2tB, wstB, wetB, embB, b1, b2, gamma, beta, bs, be,
      hsB, heB, ksB, keB, qsB, qeB);
  scan_kernel<<<32, 256, 0, stream>>>(hsB, heB, ksB, keB, qsB, qeB, rB);
  head_kernel<<<B_N, 64, 0, stream>>>(rB, Wrp, brp, Wout, bout, (float*)d_out);
}
